// OptimizedAttentionModule_55224689492135
// MI455X (gfx1250) — matrix operand order repaired and checked by reading, not compiled, not run on hardware
//
#include <hip/hip_runtime.h>
#include <hip/hip_bf16.h>
#include <math.h>

// ---------- problem constants ----------
#define BD   2
#define SD   2048
#define HDIM 1024
#define NH   16
#define HD   64
#define MTOT (BD*SD)          // 4096 rows of (b,s)

typedef __attribute__((ext_vector_type(16))) _Float16 v16h;
typedef __attribute__((ext_vector_type(8)))  _Float16 v8h;
typedef __attribute__((ext_vector_type(8)))  float    v8f;

// ---------------------------------------------------------------------
// CDNA5 async copy: 16 bytes global -> LDS, tracked by ASYNCcnt.
// Flat addr of a __shared__ object carries the raw LDS offset in
// addr[31:0] (ISA 10.2 aperture encoding), so truncation gives VDST.
// ---------------------------------------------------------------------
__device__ __forceinline__ void async_ld16(void* lds, const void* g) {
  uint32_t l = (uint32_t)(uintptr_t)lds;
  uint64_t a = (uint64_t)(uintptr_t)g;
  asm volatile("global_load_async_to_lds_b128 %0, %1, off"
               :: "v"(l), "v"(a) : "memory");
}
__device__ __forceinline__ void wait_async_le5() {
  asm volatile("s_wait_asynccnt 0x5" ::: "memory");
}
__device__ __forceinline__ void wait_async_0() {
  asm volatile("s_wait_asynccnt 0x0" ::: "memory");
}

// =====================================================================
// 1) LayerNorm: x[4096][1024] f32 -> xn f16
// =====================================================================
__global__ __launch_bounds__(256) void k_ln(const float* __restrict__ x,
                                            const float* __restrict__ gamma,
                                            const float* __restrict__ beta,
                                            _Float16* __restrict__ xn) {
  const int row = blockIdx.x;
  const int t   = threadIdx.x;
  const float* xr = x + (size_t)row * HDIM;
  float vals[4];
  float s = 0.f, ss = 0.f;
#pragma unroll
  for (int i = 0; i < 4; ++i) {
    float v = xr[t + i * 256];
    vals[i] = v; s += v; ss += v * v;
  }
#pragma unroll
  for (int m = 16; m >= 1; m >>= 1) {
    s  += __shfl_xor(s,  m, 32);
    ss += __shfl_xor(ss, m, 32);
  }
  __shared__ float rs[8], rss[8];
  if ((t & 31) == 0) { rs[t >> 5] = s; rss[t >> 5] = ss; }
  __syncthreads();
  s = 0.f; ss = 0.f;
#pragma unroll
  for (int i = 0; i < 8; ++i) { s += rs[i]; ss += rss[i]; }
  const float mu  = s * (1.0f / HDIM);
  const float var = ss * (1.0f / HDIM) - mu * mu;
  const float inv = rsqrtf(var + 1e-5f);
#pragma unroll
  for (int i = 0; i < 4; ++i) {
    int c = t + i * 256;
    float v = (vals[i] - mu) * inv * gamma[c] + beta[c];
    xn[(size_t)row * HDIM + c] = (_Float16)v;
  }
}

// =====================================================================
// 2) Weight convert f32 -> f16 (Wq,Wk,Wv,Wo stacked: 4 x 1M elements)
// =====================================================================
__global__ __launch_bounds__(256) void k_cvt(const float* __restrict__ w0,
                                             const float* __restrict__ w1,
                                             const float* __restrict__ w2,
                                             const float* __restrict__ w3,
                                             _Float16* __restrict__ dst) {
  const int which = blockIdx.y;
  const float* src = (which == 0) ? w0 : (which == 1) ? w1 : (which == 2) ? w2 : w3;
  const size_t i = (size_t)blockIdx.x * 256 + threadIdx.x;      // grid.x = 4096
  dst[(size_t)which * (HDIM * HDIM) + i] = (_Float16)src[i];
}

// ---------------------------------------------------------------------
// Fragment builders (layouts per CDNA5 ISA §7.12.2)
// ---------------------------------------------------------------------
__device__ __forceinline__ v16h make_afrag(const _Float16* rowp, int akoff) {
  v8h lo = *(const v8h*)(rowp + akoff);
  v8h hi = *(const v8h*)(rowp + 16 + akoff);
  v16h a;
#pragma unroll
  for (int i = 0; i < 8; ++i) { a[i] = lo[i]; a[i + 8] = hi[i]; }
  return a;
}
__device__ __forceinline__ v16h make_bfrag(const _Float16* rowp, int bkoff) {
  v8h lo = *(const v8h*)(rowp + bkoff);
  v8h hi = *(const v8h*)(rowp + 16 + bkoff);
  v16h b;
#pragma unroll
  for (int i = 0; i < 8; ++i) { b[i] = lo[i]; b[i + 8] = hi[i]; }
  return b;
}

#define LDS_PITCH 40   // 32 halves + 8 pad -> 80B row pitch, de-conflicts banks

// =====================================================================
// Shared GEMM core: block tile 256M x 64N, 8 waves x (32M x 64N),
// K stepped by 32, async-LDS double buffered with manually unrolled
// buffer-0 / buffer-1 phases (compile-time buffer index => immediate
// ds offsets, single accumulator version, no register rotation).
// Per stage: A-tile 256x32 f16 (16KB, 4 chunks/thread) +
//            B-tile  64x32 f16 ( 4KB, 1 chunk/thread).
// =====================================================================
struct GemmAcc { v8f a[2][4]; };

template <typename IssueFn>
__device__ __forceinline__ void gemm_mainloop(GemmAcc& acc,
                                              _Float16 (*As)[256][LDS_PITCH],
                                              _Float16 (*Bs)[64][LDS_PITCH],
                                              int wave, int lm, int akoff, int bkoff,
                                              IssueFn issue) {
  auto compute = [&](int st) {
    v16h a0 = make_afrag(&As[st][wave * 32 + lm][0], akoff);
    v16h a1 = make_afrag(&As[st][wave * 32 + 16 + lm][0], akoff);
#pragma unroll
    for (int nt = 0; nt < 4; ++nt) {
      v16h b = make_bfrag(&Bs[st][nt * 16 + lm][0], bkoff);
      acc.a[0][nt] = __builtin_amdgcn_wmma_f32_16x16x32_f16(
          false, a0, false, b, (short)0, acc.a[0][nt], false, false);
      acc.a[1][nt] = __builtin_amdgcn_wmma_f32_16x16x32_f16(
          false, a1, false, b, (short)0, acc.a[1][nt], false, false);
    }
  };

  issue(0, 0);
#pragma unroll 1
  for (int k2 = 0; k2 < 16; ++k2) {
    // phase 0: consume buffer 0 (k-step 2*k2), prefetch buffer 1
    issue(1, (2 * k2 + 1) * 32);
    wait_async_le5();
    __syncthreads();
    compute(0);
    __syncthreads();
    // phase 1: consume buffer 1, prefetch buffer 0 for next k2
    if (k2 + 1 < 16) { issue(0, (2 * k2 + 2) * 32); wait_async_le5(); }
    else             { wait_async_0(); }
    __syncthreads();
    compute(1);
    __syncthreads();
  }
}

// =====================================================================
// 3) QKV GEMM.  z=0 -> Q head-major, z=1 -> K head-major, z=2 -> V^T
// =====================================================================
__global__ __launch_bounds__(256) void k_qkv(const _Float16* __restrict__ xn,
                                             const _Float16* __restrict__ wh,
                                             _Float16* __restrict__ qh,
                                             _Float16* __restrict__ kh,
                                             _Float16* __restrict__ vt) {
  __shared__ __align__(16) _Float16 As[2][256][LDS_PITCH];
  __shared__ __align__(16) _Float16 Bs[2][64][LDS_PITCH];
  const int t = threadIdx.x;
  const int lane = t & 31, wave = t >> 5;
  const int lm = lane & 15, hi = lane >> 4;
  const int z  = blockIdx.z;
  const _Float16* W = wh + (size_t)z * (HDIM * HDIM);
  const int mblk = blockIdx.x * 256;
  const int n0   = blockIdx.y * 64;
  const int akoff = hi ? 8 : 0;
  const int bkoff = hi ? 8 : 0;

  GemmAcc acc = {};
  auto issue = [&](int st, int k0) {
#pragma unroll
    for (int c = 0; c < 4; ++c) {
      const int ar = (t + c * 256) >> 2, ao = ((t + c * 256) & 3) * 8;
      async_ld16(&As[st][ar][ao], xn + (size_t)(mblk + ar) * HDIM + k0 + ao);
    }
    const int br = t >> 2, bo = (t & 3) * 8;
    async_ld16(&Bs[st][br][bo], W + (size_t)(n0 + br) * HDIM + k0 + bo);
  };
  gemm_mainloop(acc, As, Bs, wave, lm, akoff, bkoff, issue);

  // Epilogue: scatter into head-major Q/K or transposed V, f16
#pragma unroll
  for (int half = 0; half < 2; ++half) {
#pragma unroll
    for (int nt = 0; nt < 4; ++nt) {
#pragma unroll
      for (int r = 0; r < 8; ++r) {
        const int n = n0 + nt * 16 + lm;
        const int m = mblk + wave * 32 + half * 16 + r + (hi ? 8 : 0);
        const float v = acc.a[half][nt][r];
        const int b = m >> 11, s = m & (SD - 1);
        const int h = n >> 6,  d = n & (HD - 1);
        const size_t head = (size_t)(b * NH + h);
        if (z == 0)      qh[(head * SD + s) * HD + d] = (_Float16)v;
        else if (z == 1) kh[(head * SD + s) * HD + d] = (_Float16)v;
        else             vt[(head * HD + d) * SD + s] = (_Float16)v;
      }
    }
  }
}

// =====================================================================
// 4) Flash attention: grid (B*NH=32, S/128=16), block 256.
//    Each wave owns 16 q-rows, streams all 2048 keys in chunks of 32.
//    Barrier-free: K/V per head (256KB each) is L2/WGP$ resident.
// =====================================================================
__global__ __launch_bounds__(256) void k_attn(const _Float16* __restrict__ qh,
                                              const _Float16* __restrict__ kh,
                                              const _Float16* __restrict__ vt,
                                              _Float16* __restrict__ ctx) {
  const int lane = threadIdx.x & 31, wave = threadIdx.x >> 5;
  const int lm = lane & 15, hi = lane >> 4;
  const int bh = blockIdx.x;
  const int q0 = blockIdx.y * 128 + wave * 16;
  const _Float16* Q = qh + (size_t)bh * SD * HD;
  const _Float16* K = kh + (size_t)bh * SD * HD;
  const _Float16* V = vt + (size_t)bh * HD * SD;
  const int akoff = hi ? 16 : 0;
  const int bkoff = hi ? 16 : 0;

  // Q fragments for this wave's 16 rows, pre-scaled by 1/sqrt(HD)=0.125
  v16h qf[2];
  {
    const _Float16* qr = Q + (size_t)(q0 + lm) * HD;
#pragma unroll
    for (int ks = 0; ks < 2; ++ks) {
      v8h lo = *(const v8h*)(qr + ks * 32 + akoff);
      v8h h8 = *(const v8h*)(qr + ks * 32 + 8 + akoff);
#pragma unroll
      for (int i = 0; i < 8; ++i) {
        qf[ks][i]     = lo[i] * (_Float16)0.125f;
        qf[ks][i + 8] = h8[i] * (_Float16)0.125f;
      }
    }
  }

  v8f O[4] = {};
  float Mr[8], Lr[8];
#pragma unroll
  for (int r = 0; r < 8; ++r) { Mr[r] = -1e30f; Lr[r] = 0.f; }

  __shared__ __align__(16) _Float16 pbuf[8][16][32];   // per-wave P staging

  for (int n0 = 0; n0 < SD; n0 += 32) {
    if (n0 + 32 < SD) {
      __builtin_prefetch(K + (size_t)(n0 + 32 + lm) * HD, 0, 1);
      __builtin_prefetch(V + (size_t)(lm) * SD + n0 + 32, 0, 1);
    }
    // scores: 16 q-rows x 32 keys (two 16x16 C tiles), f32
    v8f c0 = {}, c1 = {};
#pragma unroll
    for (int ks = 0; ks < 2; ++ks) {
      v16h kf0 = *(const v16h*)(K + (size_t)(n0 + lm) * HD      + ks * 32 + bkoff);
      v16h kf1 = *(const v16h*)(K + (size_t)(n0 + 16 + lm) * HD + ks * 32 + bkoff);
      c0 = __builtin_amdgcn_wmma_f32_16x16x32_f16(false, qf[ks], false, kf0,
                                                  (short)0, c0, false, false);
      c1 = __builtin_amdgcn_wmma_f32_16x16x32_f16(false, qf[ks], false, kf1,
                                                  (short)0, c1, false, false);
    }
    // online softmax per row (row m = r + 8*hi lives across the 16-lane half)
#pragma unroll
    for (int r = 0; r < 8; ++r) {
      float mx = fmaxf(c0[r], c1[r]);
#pragma unroll
      for (int sh = 1; sh < 16; sh <<= 1) mx = fmaxf(mx, __shfl_xor(mx, sh, 32));
      const float nm = fmaxf(Mr[r], mx);
      const float alpha = __expf(Mr[r] - nm);
      Mr[r] = nm;
      const float p0 = __expf(c0[r] - nm);
      const float p1 = __expf(c1[r] - nm);
      float srow = p0 + p1;
#pragma unroll
      for (int sh = 1; sh < 16; sh <<= 1) srow += __shfl_xor(srow, sh, 32);
      Lr[r] = Lr[r] * alpha + srow;
#pragma unroll
      for (int dt = 0; dt < 4; ++dt) O[dt][r] *= alpha;
      pbuf[wave][r + (hi ? 8 : 0)][lm]      = (_Float16)p0;
      pbuf[wave][r + (hi ? 8 : 0)][16 + lm] = (_Float16)p1;
    }
    asm volatile("" ::: "memory");   // keep LDS store->load program order (DS in-order per wave)
    // re-layout P (C-frag -> A-frag) through per-wave LDS
    v16h pa;
    {
      v8h lo = *(const v8h*)&pbuf[wave][lm][akoff];
      v8h h8 = *(const v8h*)&pbuf[wave][lm][8 + akoff];
#pragma unroll
      for (int i = 0; i < 8; ++i) { pa[i] = lo[i]; pa[i + 8] = h8[i]; }
    }
    // O += P(16x32) x V(32x64)   (V transposed -> B frags contiguous)
#pragma unroll
    for (int dt = 0; dt < 4; ++dt) {
      v16h vf = *(const v16h*)(V + (size_t)(dt * 16 + lm) * SD + n0 + bkoff);
      O[dt] = __builtin_amdgcn_wmma_f32_16x16x32_f16(false, pa, false, vf,
                                                     (short)0, O[dt], false, false);
    }
  }

  // normalize and write context [B,S,H] f16 (token-major for final GEMM)
  const int b = bh >> 4, h = bh & (NH - 1);
#pragma unroll
  for (int dt = 0; dt < 4; ++dt) {
#pragma unroll
    for (int r = 0; r < 8; ++r) {
      const float val = O[dt][r] / Lr[r];
      const int srow  = q0 + r + (hi ? 8 : 0);
      const int col   = h * HD + dt * 16 + lm;
      ctx[(size_t)(b * SD + srow) * HDIM + col] = (_Float16)val;
    }
  }
}

// =====================================================================
// 5) Output projection: y = ctx @ Wo^T + bo + x  (f32 out)
// =====================================================================
__global__ __launch_bounds__(256) void k_out(const _Float16* __restrict__ ctx,
                                             const _Float16* __restrict__ woh,
                                             const float* __restrict__ bo,
                                             const float* __restrict__ x,
                                             float* __restrict__ out) {
  __shared__ __align__(16) _Float16 As[2][256][LDS_PITCH];
  __shared__ __align__(16) _Float16 Bs[2][64][LDS_PITCH];
  const int t = threadIdx.x;
  const int lane = t & 31, wave = t >> 5;
  const int lm = lane & 15, hi = lane >> 4;
  const int mblk = blockIdx.x * 256;
  const int n0   = blockIdx.y * 64;
  const int akoff = hi ? 8 : 0;
  const int bkoff = hi ? 8 : 0;

  GemmAcc acc = {};
  auto issue = [&](int st, int k0) {
#pragma unroll
    for (int c = 0; c < 4; ++c) {
      const int ar = (t + c * 256) >> 2, ao = ((t + c * 256) & 3) * 8;
      async_ld16(&As[st][ar][ao], ctx + (size_t)(mblk + ar) * HDIM + k0 + ao);
    }
    const int br = t >> 2, bo8 = (t & 3) * 8;
    async_ld16(&Bs[st][br][bo8], woh + (size_t)(n0 + br) * HDIM + k0 + bo8);
  };
  gemm_mainloop(acc, As, Bs, wave, lm, akoff, bkoff, issue);

#pragma unroll
  for (int half = 0; half < 2; ++half) {
#pragma unroll
    for (int nt = 0; nt < 4; ++nt) {
#pragma unroll
      for (int r = 0; r < 8; ++r) {
        const int n = n0 + nt * 16 + lm;
        const int m = mblk + wave * 32 + half * 16 + r + (hi ? 8 : 0);
        const size_t idx = (size_t)m * HDIM + n;
        out[idx] = acc.a[half][nt][r] + bo[n] + x[idx];
      }
    }
  }
}

// =====================================================================
// launcher
// =====================================================================
extern "C" void kernel_launch(void* const* d_in, const int* in_sizes, int n_in,
                              void* d_out, int out_size, void* d_ws, size_t ws_size,
                              hipStream_t stream) {
  const float* x     = (const float*)d_in[0];
  const float* Wq    = (const float*)d_in[1];
  const float* Wk    = (const float*)d_in[2];
  const float* Wv    = (const float*)d_in[3];
  const float* Wo    = (const float*)d_in[4];
  const float* bo    = (const float*)d_in[5];
  const float* gamma = (const float*)d_in[6];
  const float* beta  = (const float*)d_in[7];
  float* out = (float*)d_out;

  char* ws = (char*)d_ws;
  const size_t MB = 1u << 20;
  _Float16* xn  = (_Float16*)(ws + 0 * MB);    // [4096][1024]          8 MB
  _Float16* wh  = (_Float16*)(ws + 8 * MB);    // 4 x [1024][1024]      8 MB
  _Float16* qh  = (_Float16*)(ws + 16 * MB);   // [32][2048][64]        8 MB
  _Float16* kh  = (_Float16*)(ws + 24 * MB);   // [32][2048][64]        8 MB
  _Float16* vt  = (_Float16*)(ws + 32 * MB);   // [32][64][2048]        8 MB
  _Float16* cx  = (_Float16*)(ws + 40 * MB);   // [4096][1024]          8 MB

  k_ln  <<<dim3(MTOT), dim3(256), 0, stream>>>(x, gamma, beta, xn);
  k_cvt <<<dim3((HDIM * HDIM) / 256, 4), dim3(256), 0, stream>>>(Wq, Wk, Wv, Wo, wh);
  k_qkv <<<dim3(MTOT / 256, HDIM / 64, 3), dim3(256), 0, stream>>>(xn, wh, qh, kh, vt);
  k_attn<<<dim3(BD * NH, SD / 128), dim3(256), 0, stream>>>(qh, kh, vt, cx);
  k_out <<<dim3(MTOT / 256, HDIM / 64), dim3(256), 0, stream>>>(
      cx, wh + (size_t)3 * HDIM * HDIM, bo, x, out);
}
